// Net_52390011077125
// MI455X (gfx1250) — compile-verified, matrix operand order checked by reading, not run on hardware
//
#include <hip/hip_runtime.h>
#include <hip/hip_bf16.h>
#include <math.h>

// ---------------------------------------------------------------------------
// Problem constants (from the reference)
// ---------------------------------------------------------------------------
#define KD   128
#define NS   4096
#define NE   4096
#define NK   128
#define ESE  131072
#define EEK  16384
#define BB   8192
#define TAU_INV 2.0f
#define ALPHAS 0.1f
#define ALPHAE 0.1f

typedef __attribute__((ext_vector_type(16))) _Float16 v16h;
typedef __attribute__((ext_vector_type(8)))  _Float16 v8h;
typedef __attribute__((ext_vector_type(8)))  float    v8f;

// fast 1-ulp reciprocal -> single v_rcp_f32
__device__ __forceinline__ float frcp(float x) { return __builtin_amdgcn_rcpf(x); }

// ---------------------------------------------------------------------------
// Order-preserving float<->uint encoding for atomic segment-max
// ---------------------------------------------------------------------------
__device__ __forceinline__ unsigned fenc(float f) {
  unsigned u = __float_as_uint(f);
  return (u & 0x80000000u) ? ~u : (u | 0x80000000u);
}
__device__ __forceinline__ float fdec(unsigned e) {
  unsigned u = (e & 0x80000000u) ? (e & 0x7FFFFFFFu) : ~e;
  return __uint_as_float(u);
}
#define NEGINF_ENC 0x007FFFFFu   // fenc(-inf); init value 0 is below this

// ---------------------------------------------------------------------------
// Row gather (float4-vectorized): out[i,:] = table[ids[i],:]
// ---------------------------------------------------------------------------
__global__ void k_gather_rows(const float4* __restrict__ tab,
                              const int* __restrict__ ids,
                              float4* __restrict__ out, int N) {
  int t = blockIdx.x * blockDim.x + threadIdx.x;   // N*32 float4s
  if (t >= N * 32) return;
  int i = t >> 5, c = t & 31;
  out[t] = tab[(long)ids[i] * 32 + c];
}

// ---------------------------------------------------------------------------
// Weight pre-pack: emit B-operand of V_WMMA_F32_16X16X32_F16 in per-lane
// order so the GEMM hot loop is two contiguous b128 loads per lane.
// Layout: Wp[((ntile*4 + j)*32 + lane)*16 + 0..15] halves.
//   transW=0 : B[k][n] = W[k*128+n]   (A @ W)
//   transW=1 : B[k][n] = W[n*128+k]   (A @ W^T)
// ---------------------------------------------------------------------------
__global__ void k_pack_w(const float* __restrict__ W,
                         _Float16* __restrict__ Wp, int transW) {
  int t = blockIdx.x * blockDim.x + threadIdx.x;   // 8*4*32 = 1024
  if (t >= 1024) return;
  int lane = t & 31, j = (t >> 5) & 3, nt = t >> 7;
  int rc = lane & 15, h = lane >> 4;
  int n = nt * 16 + rc;
  _Float16* dst = Wp + (size_t)t * 16;
#pragma unroll
  for (int v = 0; v < 8; ++v) {
    int kk = (v < 4) ? (h * 8 + 2 * v) : (16 + h * 8 + 2 * (v - 4));
    int k = j * 32 + kk;
    float w0, w1;
    if (transW) { w0 = W[n * KD + k];     w1 = W[n * KD + k + 1]; }
    else        { w0 = W[k * KD + n];     w1 = W[(k + 1) * KD + n]; }
    dst[2 * v]     = (_Float16)w0;
    dst[2 * v + 1] = (_Float16)w1;
  }
}

// ---------------------------------------------------------------------------
// WMMA GEMM:  C[M x 128] = act( f16(A[M x 128]) @ Wp (+bias) )
// A staged in LDS as f16 (shared by all 8 waves); B from pre-packed Wp.
// K-loop body: 2x ds_load_b128 + 2x global_load_b128 + 1x v_wmma.
// act: 0=none, 1=sigmoid.  grid.x = M/16, block = 256.
// ---------------------------------------------------------------------------
__global__ void k_gemm128_wmma(const float* __restrict__ A,
                               const _Float16* __restrict__ Wp,
                               float* __restrict__ C,
                               const float* __restrict__ bias, int act) {
  __shared__ _Float16 As[16 * KD] __attribute__((aligned(16)));
  const int m0 = blockIdx.x * 16;

  // cooperative A-tile stage: 2048 floats -> 2048 halves
  {
    const float4* src = (const float4*)(A + (long)m0 * KD);
    float4 x0 = src[threadIdx.x * 2];
    float4 x1 = src[threadIdx.x * 2 + 1];
    v8h hv;
    hv[0] = (_Float16)x0.x; hv[1] = (_Float16)x0.y;
    hv[2] = (_Float16)x0.z; hv[3] = (_Float16)x0.w;
    hv[4] = (_Float16)x1.x; hv[5] = (_Float16)x1.y;
    hv[6] = (_Float16)x1.z; hv[7] = (_Float16)x1.w;
    *(v8h*)&As[threadIdx.x * 8] = hv;
  }
  __syncthreads();

  const int lane = threadIdx.x & 31;
  const int wave = threadIdx.x >> 5;        // n-tile
  const int rc   = lane & 15;
  const int half = lane >> 4;
  const int n0   = wave * 16;

  v8f acc = {};
#pragma unroll
  for (int j = 0; j < 4; ++j) {
    v8h a0 = *(const v8h*)&As[rc * KD + 32 * j + half * 8];
    v8h a1 = *(const v8h*)&As[rc * KD + 32 * j + 16 + half * 8];
    const _Float16* bp = Wp + (size_t)(((wave * 4 + j) * 32 + lane) * 16);
    v8h b0 = *(const v8h*)bp;
    v8h b1 = *(const v8h*)(bp + 8);
    v16h a = __builtin_shufflevector(a0, a1, 0,1,2,3,4,5,6,7,8,9,10,11,12,13,14,15);
    v16h b = __builtin_shufflevector(b0, b1, 0,1,2,3,4,5,6,7,8,9,10,11,12,13,14,15);
    acc = __builtin_amdgcn_wmma_f32_16x16x32_f16(false, a, false, b,
                                                 (short)0, acc, false, false);
  }
  const float bv = bias ? bias[n0 + rc] : 0.0f;
#pragma unroll
  for (int r = 0; r < 8; ++r) {
    float v = acc[r] + bv;
    if (act == 1) v = frcp(1.0f + __expf(-v));   // sigmoid via v_rcp_f32
    C[(long)(m0 + r + 8 * half) * KD + n0 + rc] = v;
  }
}

// ---------------------------------------------------------------------------
// Per-node attention terms (wave per node, one float4 per lane)
// ---------------------------------------------------------------------------
__global__ void k_node_scores(const float* __restrict__ h,
                              const float* __restrict__ as_,
                              const float* __restrict__ ad_,
                              float* __restrict__ ss, float* __restrict__ sd,
                              int N) {
  int i = blockIdx.x * (blockDim.x >> 5) + (threadIdx.x >> 5);
  if (i >= N) return;
  int lane = threadIdx.x & 31;
  float4 v = ((const float4*)(h + (long)i * KD))[lane];
  float4 s = ((const float4*)as_)[lane];
  float4 d = ((const float4*)ad_)[lane];
  float s0 = v.x * s.x + v.y * s.y + v.z * s.z + v.w * s.w;
  float s1 = v.x * d.x + v.y * d.y + v.z * d.z + v.w * d.w;
#pragma unroll
  for (int o = 16; o > 0; o >>= 1) {
    s0 += __shfl_xor(s0, o, 32);
    s1 += __shfl_xor(s1, o, 32);
  }
  if (lane == 0) { ss[i] = s0; sd[i] = s1; }
}

// ---------------------------------------------------------------------------
// Segment softmax pass 1: score = lrelu(ss[a_idx] + sd[d_idx]); segmax by dst
// ---------------------------------------------------------------------------
__global__ void k_edge_score_max(const float* __restrict__ ss, const int* __restrict__ a_idx,
                                 const float* __restrict__ sd, const int* __restrict__ d_idx,
                                 float* __restrict__ sc, unsigned* __restrict__ mEnc, int E) {
  int e = blockIdx.x * blockDim.x + threadIdx.x;
  if (e >= E) return;
  float x = ss[a_idx[e]] + sd[d_idx[e]];
  float s = (x > 0.f) ? x : 0.2f * x;
  sc[e] = s;
  atomicMax(&mEnc[d_idx[e]], fenc(s));
}

__global__ void k_segmax_finalize(unsigned* __restrict__ mEnc, int N) {
  int i = blockIdx.x * blockDim.x + threadIdx.x;
  if (i >= N) return;
  unsigned e = mEnc[i];
  float m = (e <= NEGINF_ENC) ? 0.0f : fdec(e);   // empty segment -> 0 (ref behavior)
  ((float*)mEnc)[i] = m;
}

__global__ void k_edge_expsum(float* __restrict__ sc, const int* __restrict__ d_idx,
                              const float* __restrict__ m, float* __restrict__ den, int E) {
  int e = blockIdx.x * blockDim.x + threadIdx.x;
  if (e >= E) return;
  int d = d_idx[e];
  float ex = __expf(sc[e] - m[d]);
  sc[e] = ex;
  atomicAdd(&den[d], ex);
}

// One wave per edge: out[dst,:] += (ex/den[dst]) * msg[src,:]
__global__ void k_edge_agg(const float* __restrict__ sc, const int* __restrict__ d_idx,
                           const int* __restrict__ s_idx, const float* __restrict__ msg,
                           const float* __restrict__ den, float* __restrict__ out, int E) {
  int e = blockIdx.x * (blockDim.x >> 5) + (threadIdx.x >> 5);
  if (e >= E) return;
  int lane = threadIdx.x & 31;
  int d = d_idx[e];
  const float* mrow = msg + (long)s_idx[e] * KD;
  __builtin_prefetch(mrow + lane * 4, 0, 3);       // global_prefetch_b8
  float alpha = sc[e] * frcp(den[d] + 1e-16f);
  float4 mv = ((const float4*)mrow)[lane];
  float* orow = out + (long)d * KD + lane * 4;
  atomicAdd(&orow[0], alpha * mv.x);
  atomicAdd(&orow[1], alpha * mv.y);
  atomicAdd(&orow[2], alpha * mv.z);
  atomicAdd(&orow[3], alpha * mv.w);
}

__global__ void k_elu_inplace(float* __restrict__ x, int n) {
  int i = blockIdx.x * blockDim.x + threadIdx.x;
  if (i >= n) return;
  float v = x[i];
  x[i] = (v > 0.f) ? v : (__expf(v) - 1.0f);
}

// ---------------------------------------------------------------------------
// Contrastive loss helpers
// ---------------------------------------------------------------------------
__global__ void k_normalize_rows_f16(const float* __restrict__ x,
                                     _Float16* __restrict__ z, int N) {
  int i = blockIdx.x * (blockDim.x >> 5) + (threadIdx.x >> 5);
  if (i >= N) return;
  int lane = threadIdx.x & 31;
  float4 v = ((const float4*)(x + (long)i * KD))[lane];
  float s = v.x * v.x + v.y * v.y + v.z * v.z + v.w * v.w;
#pragma unroll
  for (int o = 16; o > 0; o >>= 1) s += __shfl_xor(s, o, 32);
  float inv = frcp(sqrtf(s) + 1e-12f);
  v8h* dst = (v8h*)(z + (long)i * KD);
  _Float16 hv[4] = {(_Float16)(v.x * inv), (_Float16)(v.y * inv),
                    (_Float16)(v.z * inv), (_Float16)(v.w * inv)};
  ((ulong1*)dst)[lane] = *(ulong1*)hv;   // 8-byte store of 4 halves
}

// Fused sim tile: exp(z1 z2^T / tau) reduced into rowsum / colsum / diag.
// z is f16 row-major, so WMMA operands are direct contiguous b128 loads.
__global__ void k_closs_sim_wmma(const _Float16* __restrict__ z1,
                                 const _Float16* __restrict__ z2,
                                 float* __restrict__ rowsum,
                                 float* __restrict__ colsum,
                                 float* __restrict__ diagv, int N) {
  int tile = blockIdx.x * (blockDim.x >> 5) + (threadIdx.x >> 5);
  int nt = N / 16;
  if (tile >= nt * nt) return;
  int tm = tile / nt, tn = tile % nt;
  int m0 = tm * 16, n0 = tn * 16;
  int lane = threadIdx.x & 31;
  int rc = lane & 15, half = lane >> 4;

  const _Float16* arow = z1 + (long)(m0 + rc) * KD + half * 8;
  const _Float16* brow = z2 + (long)(n0 + rc) * KD + half * 8;
  v8f acc = {};
#pragma unroll
  for (int j = 0; j < 4; ++j) {
    v8h a0 = *(const v8h*)(arow + 32 * j);
    v8h a1 = *(const v8h*)(arow + 32 * j + 16);
    v8h b0 = *(const v8h*)(brow + 32 * j);
    v8h b1 = *(const v8h*)(brow + 32 * j + 16);
    v16h a = __builtin_shufflevector(a0, a1, 0,1,2,3,4,5,6,7,8,9,10,11,12,13,14,15);
    v16h b = __builtin_shufflevector(b0, b1, 0,1,2,3,4,5,6,7,8,9,10,11,12,13,14,15);
    acc = __builtin_amdgcn_wmma_f32_16x16x32_f16(false, a, false, b,
                                                 (short)0, acc, false, false);
  }
  // acc[r] = sim(m0 + r + 8*half, n0 + rc)
  float ex[8]; float csum = 0.f;
#pragma unroll
  for (int r = 0; r < 8; ++r) {
    float e = __expf(acc[r] * TAU_INV);
    ex[r] = e; csum += e;
  }
  csum += __shfl_xor(csum, 16, 32);                   // fold the two half-waves
  if (half == 0) atomicAdd(&colsum[n0 + rc], csum);
#pragma unroll
  for (int r = 0; r < 8; ++r) {
    float v = ex[r];
    v += __shfl_xor(v, 1, 32);
    v += __shfl_xor(v, 2, 32);
    v += __shfl_xor(v, 4, 32);
    v += __shfl_xor(v, 8, 32);
    if (rc == 0) atomicAdd(&rowsum[m0 + r + 8 * half], v);
  }
  if (tm == tn) {
    int r = rc - 8 * half;
    if (r >= 0 && r < 8) diagv[m0 + rc] = acc[r] * TAU_INV;
  }
}

__global__ void k_closs_finalize(const float* __restrict__ rowsum,
                                 const float* __restrict__ colsum,
                                 const float* __restrict__ diagv,
                                 float weight, float* __restrict__ accum, int N) {
  __shared__ float sh[256];
  float s = 0.f;
  for (int i = threadIdx.x; i < N; i += blockDim.x)
    s += (__logf(rowsum[i]) - diagv[i]) + (__logf(colsum[i]) - diagv[i]);
  sh[threadIdx.x] = s;
  __syncthreads();
  for (int o = 128; o > 0; o >>= 1) {
    if ((int)threadIdx.x < o) sh[threadIdx.x] += sh[threadIdx.x + o];
    __syncthreads();
  }
  if (threadIdx.x == 0) atomicAdd(accum, weight * sh[0] / (float)N);
}

// ---------------------------------------------------------------------------
// Prediction head helpers
// ---------------------------------------------------------------------------
__global__ void k_sub(const float* __restrict__ a, const float* __restrict__ b,
                      float* __restrict__ c, int n) {
  int i = blockIdx.x * blockDim.x + threadIdx.x;
  if (i < n) c[i] = a[i] - b[i];
}

__global__ void k_weighted_mean(const float* __restrict__ o,
                                const float* __restrict__ kr,
                                float* __restrict__ out, int Bn) {
  int b = blockIdx.x * (blockDim.x >> 5) + (threadIdx.x >> 5);
  if (b >= Bn) return;
  int lane = threadIdx.x & 31;
  float4 ov = ((const float4*)(o + (long)b * KD))[lane];
  float4 wv = ((const float4*)(kr + (long)b * KD))[lane];
  float num = ov.x * wv.x + ov.y * wv.y + ov.z * wv.z + ov.w * wv.w;
  float den = wv.x + wv.y + wv.z + wv.w;
#pragma unroll
  for (int off = 16; off > 0; off >>= 1) {
    num += __shfl_xor(num, off, 32);
    den += __shfl_xor(den, off, 32);
  }
  if (lane == 0) out[b] = num * frcp(den);
}

// ---------------------------------------------------------------------------
// Host-side orchestration
// ---------------------------------------------------------------------------
struct Ws {
  float *hs, *he, *hk;
  float *ss_s, *sd_s, *ss_e, *sd_e, *ss_k, *sd_k;
  float *sc_a, *sc_b1, *sc_b2, *sc_c;
  unsigned *mEnc_s, *mEnc_e, *mEnc_k;
  float *den_s, *den_e, *den_k;
};

static void gat_layer(const float* in_s, const float* in_e, const float* in_k,
                      float* out_s, float* out_e, float* out_k,
                      const _Float16* Wp, const float* as_, const float* ad_,
                      const int* se_src, const int* se_dst,
                      const int* ek_src, const int* ek_dst,
                      const Ws& w, hipStream_t st) {
  // projections
  k_gemm128_wmma<<<NS / 16, 256, 0, st>>>(in_s, Wp, w.hs, nullptr, 0);
  k_gemm128_wmma<<<NE / 16, 256, 0, st>>>(in_e, Wp, w.he, nullptr, 0);
  k_gemm128_wmma<<<NK / 16, 256, 0, st>>>(in_k, Wp, w.hk, nullptr, 0);
  // attention terms
  k_node_scores<<<(NS + 7) / 8, 256, 0, st>>>(w.hs, as_, ad_, w.ss_s, w.sd_s, NS);
  k_node_scores<<<(NE + 7) / 8, 256, 0, st>>>(w.he, as_, ad_, w.ss_e, w.sd_e, NE);
  k_node_scores<<<(NK + 7) / 8, 256, 0, st>>>(w.hk, as_, ad_, w.ss_k, w.sd_k, NK);

  // ---- stu <- exer : dst = se_src, score = lrelu(ss_e[se_dst] + sd_s[se_src])
  hipMemsetAsync(w.mEnc_s, 0, NS * 4, st);
  hipMemsetAsync(w.den_s, 0, NS * 4, st);
  hipMemsetAsync(out_s, 0, (size_t)NS * KD * 4, st);
  k_edge_score_max<<<(ESE + 255) / 256, 256, 0, st>>>(w.ss_e, se_dst, w.sd_s, se_src,
                                                      w.sc_a, w.mEnc_s, ESE);
  k_segmax_finalize<<<(NS + 255) / 256, 256, 0, st>>>(w.mEnc_s, NS);
  k_edge_expsum<<<(ESE + 255) / 256, 256, 0, st>>>(w.sc_a, se_src, (float*)w.mEnc_s,
                                                   w.den_s, ESE);
  k_edge_agg<<<(ESE + 7) / 8, 256, 0, st>>>(w.sc_a, se_src, se_dst, w.he, w.den_s,
                                            out_s, ESE);
  k_elu_inplace<<<(NS * KD + 255) / 256, 256, 0, st>>>(out_s, NS * KD);

  // ---- exer <- {stu, k} : shared softmax over both relations, dst = exer
  hipMemsetAsync(w.mEnc_e, 0, NE * 4, st);
  hipMemsetAsync(w.den_e, 0, NE * 4, st);
  hipMemsetAsync(out_e, 0, (size_t)NE * KD * 4, st);
  k_edge_score_max<<<(ESE + 255) / 256, 256, 0, st>>>(w.ss_s, se_src, w.sd_e, se_dst,
                                                      w.sc_b1, w.mEnc_e, ESE);
  k_edge_score_max<<<(EEK + 255) / 256, 256, 0, st>>>(w.ss_k, ek_dst, w.sd_e, ek_src,
                                                      w.sc_b2, w.mEnc_e, EEK);
  k_segmax_finalize<<<(NE + 255) / 256, 256, 0, st>>>(w.mEnc_e, NE);
  k_edge_expsum<<<(ESE + 255) / 256, 256, 0, st>>>(w.sc_b1, se_dst, (float*)w.mEnc_e,
                                                   w.den_e, ESE);
  k_edge_expsum<<<(EEK + 255) / 256, 256, 0, st>>>(w.sc_b2, ek_src, (float*)w.mEnc_e,
                                                   w.den_e, EEK);
  k_edge_agg<<<(ESE + 7) / 8, 256, 0, st>>>(w.sc_b1, se_dst, se_src, w.hs, w.den_e,
                                            out_e, ESE);
  k_edge_agg<<<(EEK + 7) / 8, 256, 0, st>>>(w.sc_b2, ek_src, ek_dst, w.hk, w.den_e,
                                            out_e, EEK);
  k_elu_inplace<<<(NE * KD + 255) / 256, 256, 0, st>>>(out_e, NE * KD);

  // ---- k <- exer (skipped on layer 2, result unused downstream)
  if (out_k) {
    hipMemsetAsync(w.mEnc_k, 0, NK * 4, st);
    hipMemsetAsync(w.den_k, 0, NK * 4, st);
    hipMemsetAsync(out_k, 0, (size_t)NK * KD * 4, st);
    k_edge_score_max<<<(EEK + 255) / 256, 256, 0, st>>>(w.ss_e, ek_src, w.sd_k, ek_dst,
                                                        w.sc_c, w.mEnc_k, EEK);
    k_segmax_finalize<<<(NK + 255) / 256, 256, 0, st>>>(w.mEnc_k, NK);
    k_edge_expsum<<<(EEK + 255) / 256, 256, 0, st>>>(w.sc_c, ek_dst, (float*)w.mEnc_k,
                                                     w.den_k, EEK);
    k_edge_agg<<<(EEK + 7) / 8, 256, 0, st>>>(w.sc_c, ek_dst, ek_src, w.he, w.den_k,
                                              out_k, EEK);
    k_elu_inplace<<<(NK * KD + 255) / 256, 256, 0, st>>>(out_k, NK * KD);
  }
}

extern "C" void kernel_launch(void* const* d_in, const int* in_sizes, int n_in,
                              void* d_out, int out_size, void* d_ws, size_t ws_size,
                              hipStream_t stream) {
  (void)in_sizes; (void)n_in; (void)out_size; (void)ws_size;
  const float* E_stu  = (const float*)d_in[0];
  const float* E_exer = (const float*)d_in[1];
  const float* E_k    = (const float*)d_in[2];
  const float* W1  = (const float*)d_in[3];
  const float* a1s = (const float*)d_in[4];
  const float* a1d = (const float*)d_in[5];
  const float* W2  = (const float*)d_in[6];
  const float* a2s = (const float*)d_in[7];
  const float* a2d = (const float*)d_in[8];
  const float* pW1 = (const float*)d_in[9];
  const float* pW2 = (const float*)d_in[10];
  const float* pW3 = (const float*)d_in[11];
  const float* pb3 = (const float*)d_in[12];
  const float* kn_r = (const float*)d_in[13];
  const int* stu_ids  = (const int*)d_in[14];
  const int* exer_ids = (const int*)d_in[15];
  const int* k_ids    = (const int*)d_in[16];
  const int* se_src[3] = {(const int*)d_in[17], (const int*)d_in[21], (const int*)d_in[25]};
  const int* se_dst[3] = {(const int*)d_in[18], (const int*)d_in[22], (const int*)d_in[26]};
  const int* ek_src[3] = {(const int*)d_in[19], (const int*)d_in[23], (const int*)d_in[27]};
  const int* ek_dst[3] = {(const int*)d_in[20], (const int*)d_in[24], (const int*)d_in[28]};
  const int* stu_index  = (const int*)d_in[29];
  const int* exer_index = (const int*)d_in[30];
  float* out = (float*)d_out;

  // ---- workspace carve-up (floats; every chunk is a multiple of 128) ----
  float* p = (float*)d_ws;
  auto alloc = [&](size_t n) { float* r = p; p += n; return r; };
  float* e0s = alloc((size_t)NS * KD);
  float* e0e = alloc((size_t)NE * KD);
  float* e0k = alloc((size_t)NK * KD);
  float* l1s = alloc((size_t)NS * KD);
  float* l1e = alloc((size_t)NE * KD);
  float* l1k = alloc((size_t)NK * KD);
  Ws w;
  w.hs = alloc((size_t)NS * KD); w.he = alloc((size_t)NE * KD); w.hk = alloc((size_t)NK * KD);
  w.ss_s = alloc(NS); w.sd_s = alloc(NS);
  w.ss_e = alloc(NE); w.sd_e = alloc(NE);
  w.ss_k = alloc(NK); w.sd_k = alloc(NK);
  w.sc_a = alloc(ESE); w.sc_b1 = alloc(ESE); w.sc_b2 = alloc(EEK); w.sc_c = alloc(EEK);
  w.mEnc_s = (unsigned*)alloc(NS); w.den_s = alloc(NS);
  w.mEnc_e = (unsigned*)alloc(NE); w.den_e = alloc(NE);
  w.mEnc_k = (unsigned*)alloc(NK); w.den_k = alloc(NK);
  float* main_s = alloc((size_t)NS * KD);
  float* main_e = alloc((size_t)NE * KD);
  float* v1_s = alloc((size_t)NS * KD);
  float* v1_e = alloc((size_t)NE * KD);
  float* v2_s = alloc((size_t)NS * KD);
  float* v2_e = alloc((size_t)NE * KD);
  _Float16* z1h = (_Float16*)alloc((size_t)NS * KD / 2);
  _Float16* z2h = (_Float16*)alloc((size_t)NS * KD / 2);
  float* rowsum = alloc(NS);
  float* colsum = alloc(NS);
  float* diagv  = alloc(NS);
  float* bs   = alloc((size_t)BB * KD);
  float* be   = alloc((size_t)BB * KD);
  float* pref = alloc((size_t)BB * KD);
  float* dif  = alloc((size_t)BB * KD);
  float* obuf = alloc((size_t)BB * KD);
  // pre-packed f16 weights (16384 halves = 8192 floats each)
  _Float16* W1p = (_Float16*)alloc(8192);
  _Float16* W2p = (_Float16*)alloc(8192);
  _Float16* P1p = (_Float16*)alloc(8192);
  _Float16* P2p = (_Float16*)alloc(8192);
  _Float16* P3p = (_Float16*)alloc(8192);

  // ---- pack weights into WMMA B-operand order (once per call) ----
  k_pack_w<<<4, 256, 0, stream>>>(W1, W1p, 0);
  k_pack_w<<<4, 256, 0, stream>>>(W2, W2p, 0);
  k_pack_w<<<4, 256, 0, stream>>>(pW1, P1p, 1);
  k_pack_w<<<4, 256, 0, stream>>>(pW2, P2p, 1);
  k_pack_w<<<4, 256, 0, stream>>>(pW3, P3p, 1);

  // ---- gather initial embeddings ----
  k_gather_rows<<<(NS * 32 + 255) / 256, 256, 0, stream>>>((const float4*)E_stu, stu_ids,
                                                           (float4*)e0s, NS);
  k_gather_rows<<<(NE * 32 + 255) / 256, 256, 0, stream>>>((const float4*)E_exer, exer_ids,
                                                           (float4*)e0e, NE);
  k_gather_rows<<<(NK * 32 + 255) / 256, 256, 0, stream>>>((const float4*)E_k, k_ids,
                                                           (float4*)e0k, NK);

  // ---- three 2-layer GAT runs (g0 = main, g1 = v1, g2 = v2) ----
  float* dst_s[3] = {main_s, v1_s, v2_s};
  float* dst_e[3] = {main_e, v1_e, v2_e};
  for (int g = 0; g < 3; ++g) {
    gat_layer(e0s, e0e, e0k, l1s, l1e, l1k, W1p, a1s, a1d,
              se_src[g], se_dst[g], ek_src[g], ek_dst[g], w, stream);
    gat_layer(l1s, l1e, l1k, dst_s[g], dst_e[g], nullptr, W2p, a2s, a2d,
              se_src[g], se_dst[g], ek_src[g], ek_dst[g], w, stream);
  }

  // ---- contrastive loss (accumulated into out[BB]) ----
  hipMemsetAsync(out + BB, 0, sizeof(float), stream);
  const int nt = NS / 16;
  struct { const float *h1, *h2; float wgt; } pairs[2] = {
      {v1_s, v2_s, ALPHAS}, {v1_e, v2_e, ALPHAE}};
  for (int q = 0; q < 2; ++q) {
    k_normalize_rows_f16<<<(NS + 7) / 8, 256, 0, stream>>>(pairs[q].h1, z1h, NS);
    k_normalize_rows_f16<<<(NS + 7) / 8, 256, 0, stream>>>(pairs[q].h2, z2h, NS);
    hipMemsetAsync(rowsum, 0, NS * 4, stream);
    hipMemsetAsync(colsum, 0, NS * 4, stream);
    k_closs_sim_wmma<<<(nt * nt + 7) / 8, 256, 0, stream>>>(z1h, z2h, rowsum, colsum,
                                                            diagv, NS);
    k_closs_finalize<<<1, 256, 0, stream>>>(rowsum, colsum, diagv, pairs[q].wgt,
                                            out + BB, NS);
  }

  // ---- prediction head ----
  k_gather_rows<<<(BB * 32 + 255) / 256, 256, 0, stream>>>((const float4*)main_s, stu_index,
                                                           (float4*)bs, BB);
  k_gather_rows<<<(BB * 32 + 255) / 256, 256, 0, stream>>>((const float4*)main_e, exer_index,
                                                           (float4*)be, BB);
  k_gemm128_wmma<<<BB / 16, 256, 0, stream>>>(bs, P1p, pref, nullptr, 1);
  k_gemm128_wmma<<<BB / 16, 256, 0, stream>>>(be, P2p, dif, nullptr, 1);
  k_sub<<<(BB * KD + 255) / 256, 256, 0, stream>>>(pref, dif, pref, BB * KD);
  k_gemm128_wmma<<<BB / 16, 256, 0, stream>>>(pref, P3p, obuf, pb3, 1);
  k_weighted_mean<<<(BB + 7) / 8, 256, 0, stream>>>(obuf, kn_r, out, BB);
}
